// Step2GlobalEncoder_84189948936575
// MI455X (gfx1250) — matrix operand order repaired and checked by reading, not compiled, not run on hardware
//
#include <hip/hip_runtime.h>
#include <stdint.h>

typedef __attribute__((ext_vector_type(16))) __bf16 v16bf;
typedef __attribute__((ext_vector_type(8)))  __bf16 v8bf;
typedef __attribute__((ext_vector_type(8)))  float  v8f;

#define BM 128
#define BN 128
#define BK 32
#define LDS_STRIDE 48   // bf16 elems per LDS row: 32 data + 16 elems (32B) pad (b128 frag loads stay 16B aligned)

#define USE_ASYNC 1

#if USE_ASYNC
#define ASYNC_COPY_B128(ldsoff, gvoff, gbase)                                  \
  asm volatile("global_load_async_to_lds_b128 %0, %1, %2 offset:0"             \
               :: "v"(ldsoff), "v"(gvoff), "s"(gbase) : "memory")
#define WAIT_ASYNC(n)                                                          \
  asm volatile("s_wait_asynccnt %0" :: "i"(n) : "memory")
#endif

// ---------------------------------------------------------------------------
// fp32 -> bf16 conversion kernels (operand preparation; each output is padded
// / laid out so the GEMM needs no guards and stages with pure 16B copies).
// ---------------------------------------------------------------------------

// flat convert, n8 = #elements/8 (count must be a multiple of 8)
__global__ void cvt_bf16_flat(const float* __restrict__ in,
                              __bf16* __restrict__ out, long long n8) {
  long long i = (long long)blockIdx.x * blockDim.x + threadIdx.x;
  if (i >= n8) return;
  const float4* p = reinterpret_cast<const float4*>(in) + i * 2;
  float4 a = p[0], b = p[1];
  v8bf o;
  o[0] = (__bf16)a.x; o[1] = (__bf16)a.y; o[2] = (__bf16)a.z; o[3] = (__bf16)a.w;
  o[4] = (__bf16)b.x; o[5] = (__bf16)b.y; o[6] = (__bf16)b.z; o[7] = (__bf16)b.w;
  reinterpret_cast<v8bf*>(out)[i] = o;
}

// column-padded convert: in [R][Cin] -> out [R][Cout], zeros for c >= Cin
__global__ void cvt_bf16_pad2d(const float* __restrict__ in,
                               __bf16* __restrict__ out,
                               int R, int Cin, int Cout) {
  int idx = blockIdx.x * 256 + threadIdx.x;
  if (idx >= R * Cout) return;
  int r = idx / Cout, c = idx - r * Cout;
  out[idx] = (__bf16)((c < Cin) ? in[(size_t)r * Cin + c] : 0.0f);
}

// transpose + K-pad: in [Kin][Nn] -> out [Nn][Kout], zeros for k >= Kin
__global__ void cvt_bf16_transpose_pad(const float* __restrict__ in,
                                       __bf16* __restrict__ out,
                                       int Kin, int Kout, int Nn) {
  int idx = blockIdx.x * 256 + threadIdx.x;
  if (idx >= Nn * Kout) return;
  int n = idx / Kout, k = idx - n * Kout;
  out[idx] = (__bf16)((k < Kin) ? in[(size_t)k * Nn + n] : 0.0f);
}

// ---------------------------------------------------------------------------
// Guard-free bf16 WMMA GEMM:  C = A[M,K] @ BT[Nn,K]^T (+bias)(+relu)
//   K multiple of 32; M multiple of 128; Nn multiple of 128 (128 or 256 here).
//   OUT_MODE: 0 = fp32 row-major [M][Nn]
//             1 = bf16 row-major [M][Nn]
//             2 = bf16 transposed [Nn][M]  (contiguous v8bf stores)
// Double-buffered LDS staging via async global->LDS copies (ASYNCcnt).
// ---------------------------------------------------------------------------
template<int OUT_MODE, bool HAS_BIAS, bool RELU>
__global__ __launch_bounds__(256)
void gemm_bf16wmma(const __bf16* __restrict__ A,
                   const __bf16* __restrict__ BT,
                   const float* __restrict__ bias,
                   void* __restrict__ Cv,
                   int M, int Nn, int K)
{
  __shared__ __align__(16) __bf16 sA[2][BM * LDS_STRIDE];   // [m][k]
  __shared__ __align__(16) __bf16 sB[2][BN * LDS_STRIDE];   // [n][k]

  const int tid  = threadIdx.x;
  const int lane = tid & 31;
  const int wave = tid >> 5;
  const int wr   = wave >> 1;      // 0..3 -> 32 rows each
  const int wc   = wave & 1;       // 0..1 -> 64 cols each

  const int mBase = blockIdx.y * BM;
  const int nBase = blockIdx.x * BN;

  v8f acc[2][4] = {};

  // Stage one 128x32 bf16 tile pair into buffer `buf` (pure 16B row copies).
  auto stage = [&](int buf, int k0) {
    #pragma unroll
    for (int i = 0; i < 2; ++i) {              // A tile: 512 16B chunks / 256 thr
      int f = i * 256 + tid;
      int r = f >> 2;
      int c = (f & 3) * 8;
      uint32_t go = (uint32_t)(((size_t)(mBase + r) * K + (size_t)(k0 + c)) * 2u);
      __bf16* lp = &sA[buf][r * LDS_STRIDE + c];
#if USE_ASYNC
      uint32_t lo = (uint32_t)(uintptr_t)lp;
      ASYNC_COPY_B128(lo, go, (const void*)A);
#else
      *reinterpret_cast<v8bf*>(lp) =
          *reinterpret_cast<const v8bf*>((const char*)A + go);
#endif
    }
    #pragma unroll
    for (int i = 0; i < 2; ++i) {              // B tile (BT rows = n)
      int f = i * 256 + tid;
      int r = f >> 2;
      int c = (f & 3) * 8;
      uint32_t go = (uint32_t)(((size_t)(nBase + r) * K + (size_t)(k0 + c)) * 2u);
      __bf16* lp = &sB[buf][r * LDS_STRIDE + c];
#if USE_ASYNC
      uint32_t lo = (uint32_t)(uintptr_t)lp;
      ASYNC_COPY_B128(lo, go, (const void*)BT);
#else
      *reinterpret_cast<v8bf*>(lp) =
          *reinterpret_cast<const v8bf*>((const char*)BT + go);
#endif
    }
  };

  const int nk = K / BK;
  stage(0, 0);

  const int mA    = lane & 15;
  const int aHalf = (lane >> 4) * 8;    // A frag: lanes 0-15 K{0..7,16..23}; 16-31 K{8..15,24..31}
  const int bHalf = (lane >> 4) * 8;    // B frag: lanes 0-15 K{0..7,16..23}; 16-31 K{8..15,24..31}

  for (int kt = 0; kt < nk; ++kt) {
    const int cur = kt & 1;

    __syncthreads();                    // everyone done reading buf cur^1
    if (kt + 1 < nk) {
      stage(cur ^ 1, (kt + 1) * BK);    // prefetch next tile (async)
#if USE_ASYNC
      WAIT_ASYNC(4);                    // wait current tile; leave 4 newest in flight
#endif
    } else {
#if USE_ASYNC
      WAIT_ASYNC(0);
#endif
    }
    __syncthreads();                    // current buffer visible to all waves

    v16bf afrag[2];
    #pragma unroll
    for (int tm = 0; tm < 2; ++tm) {
      int row = wr * 32 + tm * 16 + mA;
      union { v16bf v; v8bf h[2]; } u;
      u.h[0] = *reinterpret_cast<const v8bf*>(&sA[cur][row * LDS_STRIDE + aHalf]);
      u.h[1] = *reinterpret_cast<const v8bf*>(&sA[cur][row * LDS_STRIDE + aHalf + 16]);
      afrag[tm] = u.v;
    }
    v16bf bfrag[4];
    #pragma unroll
    for (int tn = 0; tn < 4; ++tn) {
      int col = wc * 64 + tn * 16 + mA;
      union { v16bf v; v8bf h[2]; } u;
      u.h[0] = *reinterpret_cast<const v8bf*>(&sB[cur][col * LDS_STRIDE + bHalf]);
      u.h[1] = *reinterpret_cast<const v8bf*>(&sB[cur][col * LDS_STRIDE + bHalf + 16]);
      bfrag[tn] = u.v;
    }

    #pragma unroll
    for (int tm = 0; tm < 2; ++tm)
      #pragma unroll
      for (int tn = 0; tn < 4; ++tn)
        acc[tm][tn] = __builtin_amdgcn_wmma_f32_16x16x32_bf16(
            false, afrag[tm], false, bfrag[tn],
            (short)0, acc[tm][tn], false, false);
  }

  // ---- epilogue: lane holds column (lane&15), rows mhalf..mhalf+7 ----
  const int n     = lane & 15;
  const int mhalf = (lane >> 4) * 8;
  #pragma unroll
  for (int tn = 0; tn < 4; ++tn) {
    int col = nBase + wc * 64 + tn * 16 + n;
    float bv = HAS_BIAS ? bias[col] : 0.0f;
    #pragma unroll
    for (int tm = 0; tm < 2; ++tm) {
      int row0 = mBase + wr * 32 + tm * 16 + mhalf;
      if (OUT_MODE == 2) {
        v8bf o;
        #pragma unroll
        for (int r = 0; r < 8; ++r) {
          float v = acc[tm][tn][r] + bv;
          if (RELU) v = fmaxf(v, 0.0f);
          o[r] = (__bf16)v;
        }
        *reinterpret_cast<v8bf*>(
            (__bf16*)Cv + (size_t)col * M + row0) = o;       // contiguous, 16B aligned
      } else if (OUT_MODE == 1) {
        #pragma unroll
        for (int r = 0; r < 8; ++r) {
          float v = acc[tm][tn][r] + bv;
          if (RELU) v = fmaxf(v, 0.0f);
          ((__bf16*)Cv)[(size_t)(row0 + r) * Nn + col] = (__bf16)v;
        }
      } else {
        #pragma unroll
        for (int r = 0; r < 8; ++r) {
          float v = acc[tm][tn][r] + bv;
          if (RELU) v = fmaxf(v, 0.0f);
          ((float*)Cv)[(size_t)(row0 + r) * Nn + col] = v;
        }
      }
    }
  }
}

// ---------------------------------------------------------------------------
extern "C" void kernel_launch(void* const* d_in, const int* in_sizes, int n_in,
                              void* d_out, int out_size, void* d_ws, size_t ws_size,
                              hipStream_t stream)
{
  (void)in_sizes; (void)n_in; (void)out_size; (void)ws_size;

  const int N = 4096, HID = 256, OUTD = 128;
  const int dims[5] = {768, 64, 93, 256, 768};

  // bf16 workspace layout (elements)
  __bf16* adjb = reinterpret_cast<__bf16*>(d_ws);              // 4096*4096
  __bf16* xb   = adjb + (size_t)N * N;                          // 4096*768 max
  __bf16* w1t  = xb   + (size_t)N * 768;                        // 256*768 max  [HID][dp]
  __bf16* w2t  = w1t  + (size_t)HID * 768;                      // 128*256      [OUTD][HID]
  __bf16* xwt  = w2t  + (size_t)OUTD * HID;                     // 256*4096     [HID][N]
  __bf16* hbuf = xwt  + (size_t)HID * N;                        // 4096*256     [N][HID]
  __bf16* hwt  = hbuf + (size_t)N * HID;                        // 128*4096     [OUTD][N]

  for (int v = 0; v < 5; ++v) {
    const float* adj = (const float*)d_in[v * 6 + 0];
    const float* x   = (const float*)d_in[v * 6 + 1];
    const float* W1  = (const float*)d_in[v * 6 + 2];
    const float* b1  = (const float*)d_in[v * 6 + 3];
    const float* W2  = (const float*)d_in[v * 6 + 4];
    const float* b2  = (const float*)d_in[v * 6 + 5];
    float* outv = (float*)d_out + (size_t)v * N * OUTD;

    const int d  = dims[v];
    const int dp = (d + 31) & ~31;     // K padded to multiple of BK

    // --- operand prep (fp32 -> bf16, pad K, pre-transpose B operands) ---
    {
      long long n8 = (long long)N * N / 8;
      cvt_bf16_flat<<<(unsigned)((n8 + 255) / 256), 256, 0, stream>>>(adj, adjb, n8);
    }
    if ((d & 7) == 0 && d == dp) {
      long long n8 = (long long)N * d / 8;
      cvt_bf16_flat<<<(unsigned)((n8 + 255) / 256), 256, 0, stream>>>(x, xb, n8);
    } else {
      int tot = N * dp;
      cvt_bf16_pad2d<<<(tot + 255) / 256, 256, 0, stream>>>(x, xb, N, d, dp);
    }
    {
      int tot = HID * dp;    // W1 [d][HID] -> w1t [HID][dp]
      cvt_bf16_transpose_pad<<<(tot + 255) / 256, 256, 0, stream>>>(W1, w1t, d, dp, HID);
    }
    {
      int tot = OUTD * HID;  // W2 [HID][OUTD] -> w2t [OUTD][HID]
      cvt_bf16_transpose_pad<<<(tot + 255) / 256, 256, 0, stream>>>(W2, w2t, HID, HID, OUTD);
    }

    // --- 1) XW^T[HID][N] = (x @ W1)^T ---
    gemm_bf16wmma<2, false, false><<<dim3(HID / BN, N / BM), 256, 0, stream>>>(
        xb, w1t, nullptr, xwt, N, HID, dp);
    // --- 2) H[N][HID] = relu(adj @ XW + b1) ---
    gemm_bf16wmma<1, true, true><<<dim3(HID / BN, N / BM), 256, 0, stream>>>(
        adjb, xwt, b1, hbuf, N, HID, N);
    // --- 3) HW^T[OUTD][N] = (H @ W2)^T ---
    gemm_bf16wmma<2, false, false><<<dim3(OUTD / BN, N / BM), 256, 0, stream>>>(
        hbuf, w2t, nullptr, hwt, N, OUTD, HID);
    // --- 4) out[N][OUTD] = adj @ HW + b2  (fp32) ---
    gemm_bf16wmma<0, true, false><<<dim3(OUTD / BN, N / BM), 256, 0, stream>>>(
        adjb, hwt, b2, outv, N, OUTD, N);
  }
}
